// DecoderRNN_16947940950027
// MI455X (gfx1250) — compile-verified, matrix operand order checked by reading, not run on hardware
//
#include <hip/hip_runtime.h>
#include <stdint.h>

#define Bc   16
#define Sc   64
#define KGc  128
#define Tc   32
#define Ec   512
#define Rc   512
#define Vc   50257
#define VP   50304      // padded vocab row stride (786 * 64)
#define NROW 48         // B * K mixture rows
#define XDIM 1536       // E + SRC + KGD
#define NEGI -1e9f

typedef __attribute__((ext_vector_type(16))) __bf16 bf16x16;
typedef __attribute__((ext_vector_type(8)))  float  floatx8;

union Frag { uint4 u[2]; bf16x16 v; };

static __device__ __forceinline__ unsigned short f2bf(float f) {
  unsigned int u = __float_as_uint(f);
  unsigned int r = (u + 0x7FFFu + ((u >> 16) & 1u)) >> 16;
  return (unsigned short)r;
}

// ---------------- setup kernels ----------------

__global__ void k_emb_bf16(const float* __restrict__ emb,
                           unsigned short* __restrict__ embB, int n) {
  int i = blockIdx.x * 256 + threadIdx.x;
  if (i < n) embB[i] = f2bf(emb[i]);
}

__global__ void k_tgt_emb(const int* __restrict__ tgt, const float* __restrict__ emb,
                          float* __restrict__ te) {
  int i = blockIdx.x * 256 + threadIdx.x;
  if (i >= Bc * Tc * Ec) return;
  int e = i & (Ec - 1);
  int bt = i / Ec;
  te[i] = emb[(size_t)tgt[bt] * Ec + e];
}

__global__ void k_init_hidden(const float* __restrict__ ctx, const float* __restrict__ Wc,
                              const float* __restrict__ bc,
                              float* __restrict__ h0, float* __restrict__ h1) {
  int idx = blockIdx.x * 256 + threadIdx.x;   // B * (L*R) = 16 * 1024
  if (idx >= Bc * 1024) return;
  int b = idx / 1024, j = idx % 1024;
  const float4* w = (const float4*)(Wc + (size_t)j * Ec);
  const float4* x = (const float4*)(ctx + (size_t)b * Ec);
  float acc = bc[j];
  for (int i = 0; i < Ec / 4; i++) {
    float4 ww = w[i], xx = x[i];
    acc += ww.x * xx.x + ww.y * xx.y + ww.z * xx.z + ww.w * xx.w;
  }
  if (j < Rc) h0[b * Rc + j] = acc;
  else        h1[b * Rc + (j - Rc)] = acc;
}

// ---------------- per-step kernels ----------------

// attention (src + KG), builds x = [emb_t, attn_src, attn_kg], copy row (dedup scatter)
__global__ void k_attn(const float* __restrict__ h1cur, const float* __restrict__ src,
                       const int* __restrict__ src_len, const float* __restrict__ kg,
                       const int* __restrict__ kg_len, const int* __restrict__ kg_seq,
                       const float* __restrict__ tgtE, int t,
                       float* __restrict__ xbuf, float* __restrict__ copyb) {
  int b = blockIdx.x, tid = threadIdx.x;
  __shared__ float last[Rc], q[Rc], ss[Sc], sk[KGc];
  __shared__ int   seqs[KGc];
  __shared__ float red;
  for (int i = tid; i < Rc; i += 256) last[i] = h1cur[b * Rc + i];
  __syncthreads();
  if (tid < Sc) {
    const float* row = src + (size_t)(b * Sc + tid) * Ec;
    float a = 0.f;
    for (int i = 0; i < Rc; i++) a += last[i] * row[i];
    ss[tid] = (tid < src_len[b]) ? a : NEGI;
  }
  __syncthreads();
  if (tid == 0) {
    float m = -1e30f;
    for (int s = 0; s < Sc; s++) m = fmaxf(m, ss[s]);
    float sum = 0.f;
    for (int s = 0; s < Sc; s++) { ss[s] = __expf(ss[s] - m); sum += ss[s]; }
    red = 1.f / sum;
  }
  __syncthreads();
  float inv = red;
  for (int d = tid; d < Ec; d += 256) {
    float a = 0.f;
    for (int s = 0; s < Sc; s++) a += ss[s] * src[(size_t)(b * Sc + s) * Ec + d];
    a *= inv;
    q[d] = last[d] + a;
    xbuf[b * XDIM + Ec + d] = a;               // attn_src
    xbuf[b * XDIM + d] = tgtE[(size_t)(b * Tc + t) * Ec + d];  // emb_t
  }
  __syncthreads();
  if (tid < KGc) {
    const float* row = kg + (size_t)(b * KGc + tid) * Ec;
    float a = 0.f;
    for (int i = 0; i < Rc; i++) a += q[i] * row[i];
    sk[tid] = (tid < kg_len[b]) ? a : NEGI;
    seqs[tid] = kg_seq[b * KGc + tid];
  }
  __syncthreads();
  if (tid == 0) {
    float m = -1e30f;
    for (int s = 0; s < KGc; s++) m = fmaxf(m, sk[s]);
    float sum = 0.f;
    for (int s = 0; s < KGc; s++) { sk[s] = __expf(sk[s] - m); sum += sk[s]; }
    red = 1.f / sum;
  }
  __syncthreads();
  inv = red;
  for (int d = tid; d < Ec; d += 256) {
    float a = 0.f;
    for (int s = 0; s < KGc; s++) a += sk[s] * kg[(size_t)(b * KGc + s) * Ec + d];
    xbuf[b * XDIM + 2 * Ec + d] = a * inv;     // attn_kg
  }
  for (int v = tid; v < Vc; v += 256) copyb[(size_t)b * Vc + v] = 0.f;
  __syncthreads();
  // deterministic dedup scatter of a_k into the copy row (no atomics)
  if (tid < KGc) {
    int myv = seqs[tid];
    float tot = 0.f; bool first = true;
    for (int i = 0; i < KGc; i++) {
      if (seqs[i] == myv) {
        if (i < tid) { first = false; break; }
        tot += sk[i] * inv;
      }
    }
    if (first) copyb[(size_t)b * Vc + myv] = tot;
  }
}

// one GRU layer: h' = (1-z)*n + z*h ; grid = 32 blocks (2 per batch row)
__global__ void k_gru(const float* __restrict__ x, int Din, const float* __restrict__ h,
                      const float* __restrict__ Wih, const float* __restrict__ Whh,
                      const float* __restrict__ bih, const float* __restrict__ bhh,
                      float* __restrict__ hout) {
  int b = blockIdx.x >> 1;
  int j = (blockIdx.x & 1) * 256 + threadIdx.x;     // 0..511
  extern __shared__ float sm[];
  float* sx = sm;
  float* sh = sm + Din;
  for (int i = threadIdx.x; i < Din; i += 256) sx[i] = x[(size_t)b * Din + i];
  for (int i = threadIdx.x; i < Rc; i += 256)  sh[i] = h[(size_t)b * Rc + i];
  __syncthreads();
  float ir = bih[j], iz = bih[Rc + j], in = bih[2 * Rc + j];
  const float4* w0 = (const float4*)(Wih + (size_t)j * Din);
  const float4* w1 = (const float4*)(Wih + (size_t)(Rc + j) * Din);
  const float4* w2 = (const float4*)(Wih + (size_t)(2 * Rc + j) * Din);
  int D4 = Din >> 2;
  for (int i = 0; i < D4; i++) {
    float4 a = w0[i], c = w1[i], d = w2[i];
    float x0 = sx[4 * i], x1 = sx[4 * i + 1], x2 = sx[4 * i + 2], x3 = sx[4 * i + 3];
    ir += a.x * x0 + a.y * x1 + a.z * x2 + a.w * x3;
    iz += c.x * x0 + c.y * x1 + c.z * x2 + c.w * x3;
    in += d.x * x0 + d.y * x1 + d.z * x2 + d.w * x3;
  }
  float hr = bhh[j], hz = bhh[Rc + j], hn = bhh[2 * Rc + j];
  const float4* u0 = (const float4*)(Whh + (size_t)j * Rc);
  const float4* u1 = (const float4*)(Whh + (size_t)(Rc + j) * Rc);
  const float4* u2 = (const float4*)(Whh + (size_t)(2 * Rc + j) * Rc);
  for (int i = 0; i < Rc / 4; i++) {
    float4 a = u0[i], c = u1[i], d = u2[i];
    float x0 = sh[4 * i], x1 = sh[4 * i + 1], x2 = sh[4 * i + 2], x3 = sh[4 * i + 3];
    hr += a.x * x0 + a.y * x1 + a.z * x2 + a.w * x3;
    hz += c.x * x0 + c.y * x1 + c.z * x2 + c.w * x3;
    hn += d.x * x0 + d.y * x1 + d.z * x2 + d.w * x3;
  }
  float r = 1.f / (1.f + __expf(-(ir + hr)));
  float z = 1.f / (1.f + __expf(-(iz + hz)));
  float n = tanhf(in + r * hn);
  hout[(size_t)b * Rc + j] = (1.f - z) * n + z * sh[j];
}

// mixture head: prior softmax (K=3), gen gate, latent -> tanh -> bf16 A-matrix
__global__ void k_head(const float* __restrict__ rnn, const float* __restrict__ xbuf,
                       const float* __restrict__ Wp, const float* __restrict__ bp,
                       const float* __restrict__ Wl, const float* __restrict__ bl,
                       const float* __restrict__ Wg, const float* __restrict__ bg,
                       float* __restrict__ prior, float* __restrict__ genb,
                       unsigned short* __restrict__ latA) {
  int b = blockIdx.x / 6, chunk = blockIdx.x % 6;
  __shared__ float sr[Rc];
  __shared__ float pl[3];
  for (int i = threadIdx.x; i < Rc; i += 256) sr[i] = rnn[(size_t)b * Rc + i];
  __syncthreads();
  int j = chunk * 256 + threadIdx.x;   // 0..1535
  float acc = bl[j];
  const float4* w = (const float4*)(Wl + (size_t)j * Rc);
  for (int i = 0; i < Rc / 4; i++) {
    float4 ww = w[i];
    acc += ww.x * sr[4 * i] + ww.y * sr[4 * i + 1] + ww.z * sr[4 * i + 2] + ww.w * sr[4 * i + 3];
  }
  int k = j / Ec, e = j % Ec;
  latA[(size_t)(b * 3 + k) * Ec + e] = f2bf(tanhf(acc));
  if (chunk == 0 && threadIdx.x < 3) {
    float a = bp[threadIdx.x];
    const float* wp = Wp + (size_t)threadIdx.x * Rc;
    for (int i = 0; i < Rc; i++) a += wp[i] * sr[i];
    pl[threadIdx.x] = a;
  }
  if (chunk == 0 && threadIdx.x == 3) {
    float a = bg[0];
    const float* xg = xbuf + (size_t)b * XDIM;
    for (int i = 0; i < Ec; i++) a += Wg[i] * xg[i];                 // emb_t
    for (int i = 0; i < Ec; i++) a += Wg[Ec + i] * xg[2 * Ec + i];   // attn_kg
    for (int i = 0; i < Rc; i++) a += Wg[2 * Ec + i] * sr[i];        // rnn_out
    genb[b] = 1.f / (1.f + __expf(-a));
  }
  __syncthreads();
  if (chunk == 0 && threadIdx.x == 0) {
    float m = fmaxf(pl[0], fmaxf(pl[1], pl[2]));
    float e0 = __expf(pl[0] - m), e1 = __expf(pl[1] - m), e2 = __expf(pl[2] - m);
    float s = 1.f / (e0 + e1 + e2);
    prior[b * 3 + 0] = e0 * s; prior[b * 3 + 1] = e1 * s; prior[b * 3 + 2] = e2 * s;
  }
}

// vocab GEMM: scores[48][VP] = latA(48x512 bf16) @ embB(V x 512 bf16)^T, WMMA bf16.
// 8 waves/block; each wave owns 2 column tiles (32 cols) and reuses A across them:
// per k-iter = 6 A loads + 4 B loads for 6 WMMAs.
__global__ void __launch_bounds__(256) k_vgemm(const unsigned short* __restrict__ embB,
                                               const unsigned short* __restrict__ latA,
                                               float* __restrict__ scores) {
  int wave = threadIdx.x >> 5;
  int lane = threadIdx.x & 31;
  int l16  = lane & 15;          // B: column index; A: row index (within tile)
  int half = lane >> 4;
  int nbase = blockIdx.x * 256 + wave * 32;
  int n0 = nbase + l16;
  int n1 = nbase + 16 + l16;
  const unsigned short* b0 = embB + (size_t)(n0 < Vc ? n0 : (Vc - 1)) * Ec;
  const unsigned short* b1 = embB + (size_t)(n1 < Vc ? n1 : (Vc - 1)) * Ec;
  const unsigned short* a0 = latA + (size_t)(0 * 16 + l16) * Ec;
  const unsigned short* a1 = latA + (size_t)(1 * 16 + l16) * Ec;
  const unsigned short* a2 = latA + (size_t)(2 * 16 + l16) * Ec;
  floatx8 acc00 = {}, acc10 = {}, acc20 = {};
  floatx8 acc01 = {}, acc11 = {}, acc21 = {};
#pragma unroll
  for (int kb = 0; kb < Ec; kb += 32) {
    int o0 = kb + 8 * half;        // element offset of first 16B span
    Frag A0, A1, A2, B0, B1;
    A0.u[0] = *(const uint4*)(a0 + o0);  A0.u[1] = *(const uint4*)(a0 + o0 + 16);
    A1.u[0] = *(const uint4*)(a1 + o0);  A1.u[1] = *(const uint4*)(a1 + o0 + 16);
    A2.u[0] = *(const uint4*)(a2 + o0);  A2.u[1] = *(const uint4*)(a2 + o0 + 16);
    B0.u[0] = *(const uint4*)(b0 + o0);  B0.u[1] = *(const uint4*)(b0 + o0 + 16);
    B1.u[0] = *(const uint4*)(b1 + o0);  B1.u[1] = *(const uint4*)(b1 + o0 + 16);
    acc00 = __builtin_amdgcn_wmma_f32_16x16x32_bf16(false, A0.v, false, B0.v, (short)0, acc00, false, false);
    acc10 = __builtin_amdgcn_wmma_f32_16x16x32_bf16(false, A1.v, false, B0.v, (short)0, acc10, false, false);
    acc20 = __builtin_amdgcn_wmma_f32_16x16x32_bf16(false, A2.v, false, B0.v, (short)0, acc20, false, false);
    acc01 = __builtin_amdgcn_wmma_f32_16x16x32_bf16(false, A0.v, false, B1.v, (short)0, acc01, false, false);
    acc11 = __builtin_amdgcn_wmma_f32_16x16x32_bf16(false, A1.v, false, B1.v, (short)0, acc11, false, false);
    acc21 = __builtin_amdgcn_wmma_f32_16x16x32_bf16(false, A2.v, false, B1.v, (short)0, acc21, false, false);
  }
  if (n0 < Vc) {
#pragma unroll
    for (int v = 0; v < 8; v++) {
      int m = v + 8 * half;
      scores[(size_t)(0 * 16 + m) * VP + n0] = acc00[v];
      scores[(size_t)(1 * 16 + m) * VP + n0] = acc10[v];
      scores[(size_t)(2 * 16 + m) * VP + n0] = acc20[v];
    }
  }
  if (n1 < Vc) {
#pragma unroll
    for (int v = 0; v < 8; v++) {
      int m = v + 8 * half;
      scores[(size_t)(0 * 16 + m) * VP + n1] = acc01[v];
      scores[(size_t)(1 * 16 + m) * VP + n1] = acc11[v];
      scores[(size_t)(2 * 16 + m) * VP + n1] = acc21[v];
    }
  }
}

// per-row max and coef = prior/sum(exp) over vocab
__global__ void k_reduce(const float* __restrict__ scores, const float* __restrict__ prior,
                         float* __restrict__ rmax, float* __restrict__ coef) {
  int r = blockIdx.x;
  __shared__ float sm[256];
  const float* row = scores + (size_t)r * VP;
  float m = -1e30f;
  for (int v = threadIdx.x; v < Vc; v += 256) m = fmaxf(m, row[v]);
  sm[threadIdx.x] = m; __syncthreads();
  for (int s = 128; s > 0; s >>= 1) {
    if (threadIdx.x < s) sm[threadIdx.x] = fmaxf(sm[threadIdx.x], sm[threadIdx.x + s]);
    __syncthreads();
  }
  float rm = sm[0]; __syncthreads();
  float su = 0.f;
  for (int v = threadIdx.x; v < Vc; v += 256) su += __expf(row[v] - rm);
  sm[threadIdx.x] = su; __syncthreads();
  for (int s = 128; s > 0; s >>= 1) {
    if (threadIdx.x < s) sm[threadIdx.x] += sm[threadIdx.x + s];
    __syncthreads();
  }
  if (threadIdx.x == 0) { rmax[r] = rm; coef[r] = prior[r] / sm[0]; }
}

// combine mixture + copy + gate, write log-prob
__global__ void k_combine(const float* __restrict__ scores, const float* __restrict__ rmax,
                          const float* __restrict__ coef, const float* __restrict__ genb,
                          const float* __restrict__ copyb, float* __restrict__ out, int t) {
  int idx = blockIdx.x * 256 + threadIdx.x;
  if (idx >= Bc * Vc) return;
  int b = idx / Vc, v = idx % Vc;
  float pw = 0.f;
#pragma unroll
  for (int k = 0; k < 3; k++) {
    int r = b * 3 + k;
    pw += coef[r] * __expf(scores[(size_t)r * VP + v] - rmax[r]);
  }
  float g = genb[b];
  float c = g * pw + (1.f - g) * copyb[(size_t)b * Vc + v];
  out[((size_t)b * Tc + t) * Vc + v] = __logf(c);
}

// ---------------- launch ----------------

extern "C" void kernel_launch(void* const* d_in, const int* in_sizes, int n_in,
                              void* d_out, int out_size, void* d_ws, size_t ws_size,
                              hipStream_t stream) {
  (void)in_sizes; (void)n_in; (void)out_size; (void)ws_size;
  const float* src_ctx  = (const float*)d_in[0];
  const float* src_word = (const float*)d_in[1];
  const int*   src_len  = (const int*)  d_in[2];
  const float* kg_word  = (const float*)d_in[3];
  const int*   kg_len   = (const int*)  d_in[4];
  const int*   kg_seq   = (const int*)  d_in[5];
  const int*   tgt      = (const int*)  d_in[6];
  const float* emb      = (const float*)d_in[7];
  const float* Wc   = (const float*)d_in[8];
  const float* bc   = (const float*)d_in[9];
  const float* Wih0 = (const float*)d_in[10];
  const float* Whh0 = (const float*)d_in[11];
  const float* bih0 = (const float*)d_in[12];
  const float* bhh0 = (const float*)d_in[13];
  const float* Wih1 = (const float*)d_in[14];
  const float* Whh1 = (const float*)d_in[15];
  const float* bih1 = (const float*)d_in[16];
  const float* bhh1 = (const float*)d_in[17];
  const float* Wg   = (const float*)d_in[18];
  const float* bg   = (const float*)d_in[19];
  const float* Wp   = (const float*)d_in[20];
  const float* bp   = (const float*)d_in[21];
  const float* Wl   = (const float*)d_in[22];
  const float* bl   = (const float*)d_in[23];
  float* out = (float*)d_out;

  char* base = (char*)d_ws;
  size_t off = 0;
  auto take = [&](size_t bytes) -> char* {
    char* p = base + off;
    off = (off + bytes + 255) & ~(size_t)255;
    return p;
  };
  unsigned short* embB   = (unsigned short*)take((size_t)Vc * Ec * 2);
  float*          scores = (float*)take((size_t)NROW * VP * 4);
  float*          copyb  = (float*)take((size_t)Bc * Vc * 4);
  float*          tgtE   = (float*)take((size_t)Bc * Tc * Ec * 4);
  float*          h0a    = (float*)take(Bc * Rc * 4);
  float*          h0b    = (float*)take(Bc * Rc * 4);
  float*          h1a    = (float*)take(Bc * Rc * 4);
  float*          h1b    = (float*)take(Bc * Rc * 4);
  float*          xbuf   = (float*)take(Bc * XDIM * 4);
  unsigned short* latA   = (unsigned short*)take((size_t)NROW * Ec * 2);
  float*          prior  = (float*)take(Bc * 3 * 4);
  float*          genb   = (float*)take(Bc * 4);
  float*          rmax   = (float*)take(NROW * 4);
  float*          coef   = (float*)take(NROW * 4);

  int nEmb = Vc * Ec;
  k_emb_bf16<<<(nEmb + 255) / 256, 256, 0, stream>>>(emb, embB, nEmb);
  k_tgt_emb<<<(Bc * Tc * Ec + 255) / 256, 256, 0, stream>>>(tgt, emb, tgtE);
  k_init_hidden<<<(Bc * 2 * Rc + 255) / 256, 256, 0, stream>>>(src_ctx, Wc, bc, h0a, h1a);

  const int vgemm_blocks = (Vc + 255) / 256;   // 197 blocks * 256 cols
  for (int t = 0; t < Tc; t++) {
    float* h0c = (t & 1) ? h0b : h0a;  float* h0n = (t & 1) ? h0a : h0b;
    float* h1c = (t & 1) ? h1b : h1a;  float* h1n = (t & 1) ? h1a : h1b;
    k_attn<<<Bc, 256, 0, stream>>>(h1c, src_word, src_len, kg_word, kg_len, kg_seq,
                                   tgtE, t, xbuf, copyb);
    k_gru<<<Bc * 2, 256, (XDIM + Rc) * 4, stream>>>(xbuf, XDIM, h0c, Wih0, Whh0, bih0, bhh0, h0n);
    k_gru<<<Bc * 2, 256, (Rc + Rc) * 4, stream>>>(h0n, Rc, h1c, Wih1, Whh1, bih1, bhh1, h1n);
    k_head<<<Bc * 6, 256, 0, stream>>>(h1n, xbuf, Wp, bp, Wl, bl, Wg, bg, prior, genb, latA);
    k_vgemm<<<vgemm_blocks, 256, 0, stream>>>(embB, latA, scores);
    k_reduce<<<NROW, 256, 0, stream>>>(scores, prior, rmax, coef);
    k_combine<<<(Bc * Vc + 255) / 256, 256, 0, stream>>>(scores, rmax, coef, genb,
                                                         copyb, out, t);
  }
}
